// CrossAttention_63634235457562
// MI455X (gfx1250) — matrix operand order repaired and checked by reading, not compiled, not run on hardware
//
#include <hip/hip_runtime.h>
#include <hip/hip_bf16.h>

typedef __attribute__((ext_vector_type(16))) _Float16 v16h;
typedef __attribute__((ext_vector_type(8)))  _Float16 v8h;
typedef __attribute__((ext_vector_type(8)))  float    v8f;

#define NB   8
#define NC   64
#define HWL  4096
#define DD   256

__device__ __forceinline__ v16h cat8(v8h lo, v8h hi) {
  return __builtin_shufflevector(lo, hi, 0,1,2,3,4,5,6,7,8,9,10,11,12,13,14,15);
}
// A fragment (16x32 f16): p = this lane's row start (32 halfs), h = lane>>4
__device__ __forceinline__ v16h loadA(const _Float16* p, int h) {
  v8h lo = *(const v8h*)(p + h * 16);
  v8h hi = *(const v8h*)(p + h * 16 + 8);
  return cat8(lo, hi);
}
// B fragment (32x16 f16): p = this lane's 16 contiguous halfs (already +h*16)
__device__ __forceinline__ v16h loadB(const _Float16* p) {
  v8h lo = *(const v8h*)(p);
  v8h hi = *(const v8h*)(p + 8);
  return cat8(lo, hi);
}
__device__ __forceinline__ v8f zero8() {
  return v8f{0.f,0.f,0.f,0.f,0.f,0.f,0.f,0.f};
}

// Async DMA: copy 64 contiguous bytes global -> LDS for this lane.
// ISA 08_async_tensor §4: VDST = per-lane LDS byte offset, VADDR = 64-bit
// global address, INST_OFFSET added to both sides. Tracked by ASYNCcnt.
__device__ __forceinline__ void async_copy64B(unsigned lds_off, const _Float16* gsrc) {
  asm volatile(
      "global_load_async_to_lds_b128 %0, %1, off\n\t"
      "global_load_async_to_lds_b128 %0, %1, off offset:16\n\t"
      "global_load_async_to_lds_b128 %0, %1, off offset:32\n\t"
      "global_load_async_to_lds_b128 %0, %1, off offset:48"
      :: "v"(lds_off), "v"(gsrc) : "memory");
}
__device__ __forceinline__ void wait_async0() {
  asm volatile("s_wait_asynccnt 0x0" ::: "memory");
}
// Flat -> LDS byte offset (flat LDS addressing uses addr[31:0], ISA §10.2).
__device__ __forceinline__ unsigned lds_off32(const void* p) {
  return (unsigned)(unsigned long long)p;
}

// ---------------------------------------------------------------------------
// Q/K projection -> dst[b, p, d] f16.  GEMM oriented M=d, N=pixel so the
// C/D tile rows (r+8h) are contiguous d -> 16-byte stores.
// ---------------------------------------------------------------------------
__global__ __launch_bounds__(128) void proj_qk_kernel(
    const float* __restrict__ src, const float* __restrict__ W,
    const float* __restrict__ bias, _Float16* __restrict__ dst, float scale) {
  const int lane = threadIdx.x & 31;
  const int wave = threadIdx.x >> 5;
  const int n = lane & 15, h = lane >> 4;
  const int b  = blockIdx.x >> 8;
  const int p0 = (blockIdx.x & 255) << 4;

  v8f acc[4];
#pragma unroll
  for (int dt = 0; dt < 4; ++dt) acc[dt] = zero8();

#pragma unroll
  for (int s = 0; s < 2; ++s) {
    v16h bb;  // B[K=c][N=p] = src[b][s*32+K][p0+N]
    const float* bp = src + (((size_t)b * NC + s * 32) * HWL) + p0 + n;
#pragma unroll
    for (int j = 0; j < 16; ++j) bb[j] = (_Float16)bp[((j & 8) * 2 + h * 8 + (j & 7)) * HWL];
#pragma unroll
    for (int dt = 0; dt < 4; ++dt) {
      const int d0 = (wave * 4 + dt) * 16;
      const float* ap = W + (size_t)(d0 + n) * NC + s * 32;  // A[M=d][K=c]
      v16h aa;
#pragma unroll
      for (int j = 0; j < 8; ++j) aa[j]     = (_Float16)ap[h * 8 + j];
#pragma unroll
      for (int j = 0; j < 8; ++j) aa[8 + j] = (_Float16)ap[16 + h * 8 + j];
      acc[dt] = __builtin_amdgcn_wmma_f32_16x16x32_f16(
          false, aa, false, bb, (short)0, acc[dt], false, false);
    }
  }
#pragma unroll
  for (int dt = 0; dt < 4; ++dt) {
    const int d0 = (wave * 4 + dt) * 16;
    v8h o;
#pragma unroll
    for (int r = 0; r < 8; ++r)
      o[r] = (_Float16)((acc[dt][r] + bias[d0 + r + 8 * h]) * scale);
    *(v8h*)(dst + ((size_t)b * HWL + p0 + n) * DD + d0 + 8 * h) = o;
  }
}

// ---------------------------------------------------------------------------
// V projection -> dst[b, d, p] f16 (pre-transposed). M=pixel, N=d.
// ---------------------------------------------------------------------------
__global__ __launch_bounds__(128) void proj_v_kernel(
    const float* __restrict__ src, const float* __restrict__ W,
    const float* __restrict__ bias, _Float16* __restrict__ dst) {
  const int lane = threadIdx.x & 31;
  const int wave = threadIdx.x >> 5;
  const int n = lane & 15, h = lane >> 4;
  const int b  = blockIdx.x >> 8;
  const int p0 = (blockIdx.x & 255) << 4;

  v8f acc[4];
#pragma unroll
  for (int dt = 0; dt < 4; ++dt) acc[dt] = zero8();

#pragma unroll
  for (int s = 0; s < 2; ++s) {
    v16h aa;  // A[M=p][K=c] = src[b][s*32+K][p0+M]
    const float* ap = src + (((size_t)b * NC + s * 32) * HWL) + p0 + n;
#pragma unroll
    for (int j = 0; j < 8; ++j) aa[j]     = (_Float16)ap[(h * 16 + j) * HWL];
#pragma unroll
    for (int j = 0; j < 8; ++j) aa[8 + j] = (_Float16)ap[(h * 16 + 8 + j) * HWL];
#pragma unroll
    for (int dt = 0; dt < 4; ++dt) {
      const int d0 = (wave * 4 + dt) * 16;
      const float* bp = W + (size_t)(d0 + n) * NC + s * 32 + h * 16;  // B[K=c][N=d]
      v16h bb;
#pragma unroll
      for (int j = 0; j < 16; ++j) bb[j] = (_Float16)bp[j];
      acc[dt] = __builtin_amdgcn_wmma_f32_16x16x32_f16(
          false, aa, false, bb, (short)0, acc[dt], false, false);
    }
  }
#pragma unroll
  for (int dt = 0; dt < 4; ++dt) {
    const int d0 = (wave * 4 + dt) * 16;
    const float bs = bias[d0 + n];
    v8h o;
#pragma unroll
    for (int r = 0; r < 8; ++r) o[r] = (_Float16)(acc[dt][r] + bs);
    *(v8h*)(dst + ((size_t)b * DD + d0 + n) * HWL + p0 + 8 * h) = o;
  }
}

// ---------------------------------------------------------------------------
// Flash attention. 256 threads (8 waves) own (b, 128 queries); wave w owns 16
// queries with full-D fp32 accumulators. K/V tiles (32 keys) stream through
// double-buffered LDS filled by async global->LDS DMA (ASYNCcnt pipelined).
// ---------------------------------------------------------------------------
__global__ __launch_bounds__(256) void attn_kernel(
    const _Float16* __restrict__ Qh, const _Float16* __restrict__ Kh,
    const _Float16* __restrict__ Vt, float* __restrict__ out) {
  __shared__ __align__(16) _Float16 Qlds[128 * DD];     // 64 KB, [q][d]
  __shared__ __align__(16) _Float16 Klds[2][32 * DD];   // 2x16 KB, [key][d]
  __shared__ __align__(16) _Float16 Vlds[2][DD * 32];   // 2x16 KB, [d][key]
  __shared__ __align__(16) _Float16 Plds[8][16 * 32];   // 8 KB, per-wave [q][key]

  const int t = threadIdx.x;
  const int lane = t & 31;
  const int wave = t >> 5;
  const int n = lane & 15, h = lane >> 4;
  const int b   = blockIdx.x >> 5;
  const int q0b = (blockIdx.x & 31) * 128;

  // Stage Q block (128 x 256 f16) into LDS, coalesced.
  {
    const uint4* src = reinterpret_cast<const uint4*>(Qh + ((size_t)b * HWL + q0b) * DD);
    uint4* dst = reinterpret_cast<uint4*>(Qlds);
#pragma unroll
    for (int i = 0; i < 16; ++i) dst[i * 256 + t] = src[i * 256 + t];
  }

  v8f Oacc[16];
#pragma unroll
  for (int nt = 0; nt < 16; ++nt) Oacc[nt] = zero8();
  float m_run[8], l_run[8];
#pragma unroll
  for (int r = 0; r < 8; ++r) { m_run[r] = -1e30f; l_run[r] = 0.f; }

  const _Float16* Kg = Kh + (size_t)b * HWL * DD;   // [key][d]
  const _Float16* Vg = Vt + (size_t)b * DD * HWL;   // [d][key]

  // Async-DMA one 32-key tile into LDS buffer `buf`.
  // K: thread t -> key t>>3, d-chunk (t&7)*32 (64 B). V: thread t -> row d=t (64 B).
  auto issue_tile = [&](int kb, int buf) {
    async_copy64B(lds_off32(Klds[buf]) + (unsigned)t * 64u,
                  Kg + (size_t)kb * 32 * DD + (size_t)t * 32);
    async_copy64B(lds_off32(Vlds[buf]) + (unsigned)t * 64u,
                  Vg + (size_t)t * HWL + (size_t)kb * 32);
  };

  issue_tile(0, 0);

  for (int kb = 0; kb < HWL / 32; ++kb) {
    const int cur = kb & 1, nxt = cur ^ 1;
    wait_async0();                 // my DMA into tile[cur] has landed in LDS
    __syncthreads();               // everyone's DMA visible; reads of [nxt] done
    if (kb + 1 < HWL / 32) issue_tile(kb + 1, nxt);   // DMA overlaps compute

    const _Float16* kt = Klds[cur];
    const _Float16* vt = Vlds[cur];

    // S = Q * K^T : two 16x16 key-column tiles
    v8f S0 = zero8(), S1 = zero8();
#pragma unroll
    for (int dstep = 0; dstep < 8; ++dstep) {
      v16h a  = loadA(Qlds + (wave * 16 + n) * DD + dstep * 32, h);
      v16h b0 = loadB(kt + n * DD + dstep * 32 + h * 16);
      v16h b1 = loadB(kt + (16 + n) * DD + dstep * 32 + h * 16);
      S0 = __builtin_amdgcn_wmma_f32_16x16x32_f16(false, a, false, b0, (short)0, S0, false, false);
      S1 = __builtin_amdgcn_wmma_f32_16x16x32_f16(false, a, false, b1, (short)0, S1, false, false);
    }

    // Online softmax: lane holds rows {r+8h}, col n; row spans 16 lanes.
    float p0v[8], p1v[8], alpha[8];
#pragma unroll
    for (int r = 0; r < 8; ++r) {
      float s0 = S0[r], s1 = S1[r];
      float mt = fmaxf(s0, s1);
#pragma unroll
      for (int msk = 1; msk <= 8; msk <<= 1) mt = fmaxf(mt, __shfl_xor(mt, msk, 32));
      const float mnew = fmaxf(m_run[r], mt);
      const float e0 = __expf(s0 - mnew), e1 = __expf(s1 - mnew);
      float rs = e0 + e1;
#pragma unroll
      for (int msk = 1; msk <= 8; msk <<= 1) rs += __shfl_xor(rs, msk, 32);
      const float al = __expf(m_run[r] - mnew);
      l_run[r] = l_run[r] * al + rs;
      m_run[r] = mnew;
      alpha[r] = al;
      p0v[r] = e0; p1v[r] = e1;
    }
#pragma unroll
    for (int nt = 0; nt < 16; ++nt)
#pragma unroll
      for (int r = 0; r < 8; ++r) Oacc[nt][r] *= alpha[r];

    // Relayout P (C/D layout) -> A fragment via per-wave LDS scratch.
    _Float16* pl = Plds[wave];
#pragma unroll
    for (int r = 0; r < 8; ++r) {
      pl[(r + 8 * h) * 32 + n]      = (_Float16)p0v[r];
      pl[(r + 8 * h) * 32 + 16 + n] = (_Float16)p1v[r];
    }
    __syncthreads();
    v16h pa = loadA(pl + n * 32, h);

    // O += P * V : B[K=key][N=d] = Vlds[d][key]
#pragma unroll
    for (int nt = 0; nt < 16; ++nt) {
      v16h vb = loadB(vt + (nt * 16 + n) * 32 + h * 16);
      Oacc[nt] = __builtin_amdgcn_wmma_f32_16x16x32_f16(
          false, pa, false, vb, (short)0, Oacc[nt], false, false);
    }
  }

  // Epilogue: O /= l; flat [B,HW,D] == reference reshape(B,D,H,W).
  float inv[8];
#pragma unroll
  for (int r = 0; r < 8; ++r) inv[r] = 1.0f / l_run[r];
#pragma unroll
  for (int nt = 0; nt < 16; ++nt)
#pragma unroll
    for (int r = 0; r < 8; ++r) {
      const int q = q0b + wave * 16 + r + 8 * h;
      out[((size_t)b * HWL + q) * DD + nt * 16 + n] = Oacc[nt][r] * inv[r];
    }
}

// ---------------------------------------------------------------------------
extern "C" void kernel_launch(void* const* d_in, const int* in_sizes, int n_in,
                              void* d_out, int out_size, void* d_ws, size_t ws_size,
                              hipStream_t stream) {
  const float* Jp = (const float*)d_in[0];
  const float* Jg = (const float*)d_in[1];
  const float* Wq = (const float*)d_in[2];
  const float* bq = (const float*)d_in[3];
  const float* Wk = (const float*)d_in[4];
  const float* bk = (const float*)d_in[5];
  const float* Wv = (const float*)d_in[6];
  const float* bv = (const float*)d_in[7];
  float* out = (float*)d_out;

  const size_t elems = (size_t)NB * HWL * DD;
  _Float16* Qh = (_Float16*)d_ws;          // [B, HW, D]
  _Float16* Kh = Qh + elems;               // [B, HW, D]
  _Float16* Vt = Kh + elems;               // [B, D, HW] (transposed)

  const float qscale = 0.0625f;            // 1/sqrt(D) folded into Q
  dim3 pgrid(NB * (HWL / 16));             // 2048 blocks
  proj_qk_kernel<<<pgrid, 128, 0, stream>>>(Jp, Wq, bq, Qh, qscale);
  proj_qk_kernel<<<pgrid, 128, 0, stream>>>(Jg, Wk, bk, Kh, 1.0f);
  proj_v_kernel <<<pgrid, 128, 0, stream>>>(Jg, Wv, bv, Vt);

  dim3 agrid(NB * (HWL / 128));            // 256 blocks, 8 waves each
  attn_kernel<<<agrid, 256, 0, stream>>>(Qh, Kh, Vt, out);
}
